// STGNN_AutoEncoder_70171175682621
// MI455X (gfx1250) — matrix operand order repaired and checked by reading, not compiled, not run on hardware
//
#include <hip/hip_runtime.h>
#include <hip/hip_bf16.h>
#include <math.h>

// ---------------------------------------------------------------------------
// STGNN autoencoder for MI455X (gfx1250, wave32, WMMA).
// Dense GEMMs -> v_wmma_f32_16x16x32_f16, f32 accumulate. Weights pre-packed
// to a contiguous-k (k = 16g+e) B-fragment layout, staged per-workgroup in LDS (ds_load_b128),
// A loaded as float4 runs. Edge softmax -> 3-pass scatter with f32 atomics.
// ---------------------------------------------------------------------------

#define N_NODES 50000
#define N_EDGES 800000
#define N_DIM   128
#define E_DIM   32
#define OUT_DIM 64
#define HEADS   4
#define NEG_SLOPE 0.2f

typedef __attribute__((ext_vector_type(16))) _Float16 v16h;
typedef __attribute__((ext_vector_type(8)))  float    v8f;

// ---------------- B packing: f32 [K,N] -> f16 fragment order ---------------
// Bp[((ct*(K/32) + kt)*32 + lane)*16 + e] = B[k,n]
//   n = ct*16 + lane%16 ; g = lane/16 ; k = kt*32 + e + 16*g
__global__ void k_packB(const float* __restrict__ B, _Float16* __restrict__ Bp,
                        int K, int N) {
  int i = blockIdx.x * blockDim.x + threadIdx.x;
  if (i >= K * N) return;
  int e    = i & 15;
  int lane = (i >> 4) & 31;
  int kt   = (i >> 9) % (K >> 5);
  int ct   = (i >> 9) / (K >> 5);
  int n = ct * 16 + (lane & 15);
  int g = lane >> 4;
  int k = kt * 32 + e + 16 * g;
  Bp[i] = (_Float16)B[(size_t)k * N + n];
}

// ---------------- WMMA GEMM: C[M,N] = A[M,K] * Bpacked ---------------------
// M%16==0, K%32==0 (K<=128), N%64==0 at every call site.
// 128 threads = 4 waves; each wave computes a 16x64 tile. The packed B panel
// for this block's 64 columns is contiguous: staged once into LDS.
__global__ __launch_bounds__(128)
void gemm_wmma_f16(const float* __restrict__ A, const _Float16* __restrict__ Bp,
                   float* __restrict__ C, int M, int K, int N) {
  __shared__ __align__(16) _Float16 sB[4 * 4 * 512];   // KT<=4 -> <=16KB
  const int KT   = K >> 5;
  const int tid  = threadIdx.x;
  const int lane = tid & 31;
  const int wave = tid >> 5;
  const int g    = lane >> 4;
  const int l    = lane & 15;
  const int row0 = blockIdx.x * 64 + wave * 16;
  const int col0 = blockIdx.y * 64;

  // cooperative stage of the 64-column packed panel (contiguous in Bp)
  {
    const uint4* s4 = (const uint4*)(Bp + (size_t)blockIdx.y * 4 * KT * 512);
    uint4* d4 = (uint4*)sB;
    const int totalVec = (4 * KT * 512) >> 3;           // 8 halves per 16B
    for (int i = tid; i < totalVec; i += 128) d4[i] = s4[i];
  }
  __syncthreads();
  if (row0 >= M) return;

  v8f acc0 = {}, acc1 = {}, acc2 = {}, acc3 = {};
  const float* Arow = A + (size_t)(row0 + l) * K;

  for (int kt = 0; kt < KT; ++kt) {
    // A fragment: two contiguous 8-float runs -> four float4 loads
    const float4* Af = (const float4*)(Arow + kt * 32 + 16 * g);
    float4 p0 = Af[0], p1 = Af[1];      // k = kb+16g   .. +7  -> e 0..7
    float4 p2 = Af[2], p3 = Af[3];      // k = kb+16g+8 .. +7  -> e 8..15
    v16h a;
    a[0]  = (_Float16)p0.x; a[1]  = (_Float16)p0.y; a[2]  = (_Float16)p0.z; a[3]  = (_Float16)p0.w;
    a[4]  = (_Float16)p1.x; a[5]  = (_Float16)p1.y; a[6]  = (_Float16)p1.z; a[7]  = (_Float16)p1.w;
    a[8]  = (_Float16)p2.x; a[9]  = (_Float16)p2.y; a[10] = (_Float16)p2.z; a[11] = (_Float16)p2.w;
    a[12] = (_Float16)p3.x; a[13] = (_Float16)p3.y; a[14] = (_Float16)p3.z; a[15] = (_Float16)p3.w;
    // all four B fragments from LDS, then four WMMAs
    const v16h b0 = *(const v16h*)(sB + (((0 * KT + kt) * 32 + lane) << 4));
    const v16h b1 = *(const v16h*)(sB + (((1 * KT + kt) * 32 + lane) << 4));
    const v16h b2 = *(const v16h*)(sB + (((2 * KT + kt) * 32 + lane) << 4));
    const v16h b3 = *(const v16h*)(sB + (((3 * KT + kt) * 32 + lane) << 4));
    acc0 = __builtin_amdgcn_wmma_f32_16x16x32_f16(false, a, false, b0, (short)0, acc0, false, false);
    acc1 = __builtin_amdgcn_wmma_f32_16x16x32_f16(false, a, false, b1, (short)0, acc1, false, false);
    acc2 = __builtin_amdgcn_wmma_f32_16x16x32_f16(false, a, false, b2, (short)0, acc2, false, false);
    acc3 = __builtin_amdgcn_wmma_f32_16x16x32_f16(false, a, false, b3, (short)0, acc3, false, false);
  }
#pragma unroll
  for (int i = 0; i < 8; ++i) {
    float* Crow = C + (size_t)(row0 + i + 8 * g) * N + col0 + l;
    Crow[0]  = acc0[i];
    Crow[16] = acc1[i];
    Crow[32] = acc2[i];
    Crow[48] = acc3[i];
  }
}

// ---------------- small elementwise / attention kernels --------------------

__global__ void k_fill(float* p, float v, int n) {
  int i = blockIdx.x * blockDim.x + threadIdx.x;
  if (i < n) p[i] = v;
}

__device__ inline void atomicMaxF(float* addr, float val) {
  unsigned int* a = (unsigned int*)addr;
  unsigned int old = *a;
  while (__uint_as_float(old) < val) {
    unsigned int assumed = old;
    old = atomicCAS(a, assumed, __float_as_uint(val));
    if (old == assumed) break;
  }
}

// vee[k,h] = sum_d We[k,h,d] * ae[h,d]
__global__ void k_vee(const float* __restrict__ We, const float* __restrict__ ae,
                      float* __restrict__ vee, int H, int dh) {
  int i = blockIdx.x * blockDim.x + threadIdx.x;
  if (i >= E_DIM * H) return;
  int k = i / H, h = i % H;
  float s = 0.f;
  for (int d = 0; d < dh; ++d) s += We[(size_t)k * H * dh + h * dh + d] * ae[h * dh + d];
  vee[i] = s;
}

// el[n,h] = h[n,h,:].al[h,:], er likewise
__global__ void k_node_scores(const float* __restrict__ hfeat,
                              const float* __restrict__ al, const float* __restrict__ ar,
                              float* __restrict__ el, float* __restrict__ er,
                              int Nn, int H, int dh) {
  int i = blockIdx.x * blockDim.x + threadIdx.x;
  if (i >= Nn * H) return;
  int n = i / H, h = i % H;
  const float* hp = hfeat + (size_t)n * H * dh + h * dh;
  float sl = 0.f, sr = 0.f;
  for (int d = 0; d < dh; ++d) { float v = hp[d]; sl += v * al[h * dh + d]; sr += v * ar[h * dh + d]; }
  el[i] = sl; er[i] = sr;
}

// ee[e,h] = eattr[e,:] . vee[:,h]   (one thread per edge, all heads)
__global__ void k_ee(const float* __restrict__ ea, const float* __restrict__ vee,
                     float* __restrict__ ee, int H) {
  int e = blockIdx.x * blockDim.x + threadIdx.x;
  if (e >= N_EDGES) return;
  const float4* row = (const float4*)(ea + (size_t)e * E_DIM);
  float s[HEADS];
  for (int h = 0; h < H; ++h) s[h] = 0.f;
#pragma unroll
  for (int q = 0; q < E_DIM / 4; ++q) {
    float4 v = row[q];
    for (int h = 0; h < H; ++h) {
      s[h] += v.x * vee[(4 * q + 0) * H + h] + v.y * vee[(4 * q + 1) * H + h]
            + v.z * vee[(4 * q + 2) * H + h] + v.w * vee[(4 * q + 3) * H + h];
    }
  }
  for (int h = 0; h < H; ++h) ee[e * H + h] = s[h];
}

__global__ void k_edge_score(const int* __restrict__ src, const int* __restrict__ dst,
                             const float* __restrict__ el, const float* __restrict__ er,
                             const float* __restrict__ ee, float* __restrict__ esc,
                             float* __restrict__ m, int H) {
  int i = blockIdx.x * blockDim.x + threadIdx.x;
  if (i >= N_EDGES * H) return;
  int e = i / H, h = i % H;
  int s = src[e], d = dst[e];
  float v = el[s * H + h] + er[d * H + h] + ee[i];
  v = (v > 0.f) ? v : NEG_SLOPE * v;
  esc[i] = v;
  atomicMaxF(&m[d * H + h], v);
}

__global__ void k_clamp_m(float* m, int n) {
  int i = blockIdx.x * blockDim.x + threadIdx.x;
  if (i >= n) return;
  float v = m[i];
  if (!(v > -1e37f)) m[i] = 0.f;   // -inf -> 0 (matches where(isfinite(m), m, 0))
}

__global__ void k_edge_exp(const int* __restrict__ dst, const float* __restrict__ m,
                           float* __restrict__ esc, float* __restrict__ den, int H) {
  int i = blockIdx.x * blockDim.x + threadIdx.x;
  if (i >= N_EDGES * H) return;
  int e = i / H, h = i % H;
  float ex = __expf(esc[i] - m[dst[e] * H + h]);
  esc[i] = ex;
  atomicAdd(&den[dst[e] * H + h], ex);
}

__global__ void k_edge_scatter(const int* __restrict__ src, const int* __restrict__ dst,
                               const float* __restrict__ esc, const float* __restrict__ den,
                               const float* __restrict__ hfeat, float* __restrict__ out,
                               int H, int dh) {
  int F = H * dh;
  long long i = (long long)blockIdx.x * blockDim.x + threadIdx.x;
  if (i >= (long long)N_EDGES * F) return;
  int e = (int)(i / F), f = (int)(i % F);
  int h = f / dh;
  int sN = src[e], dN = dst[e];
  float a = esc[e * H + h] / (den[dN * H + h] + 1e-9f);
  atomicAdd(&out[(size_t)dN * F + f], a * hfeat[(size_t)sN * F + f]);
}

// out = [relu]( agg + res ), then optional LayerNorm with g,b
__global__ void k_gat_epilogue(const float* __restrict__ agg, const float* __restrict__ res,
                               const float* __restrict__ gam, const float* __restrict__ bet,
                               float* __restrict__ out, int Nn, int F, int act, int norm) {
  int n = blockIdx.x * blockDim.x + threadIdx.x;
  if (n >= Nn) return;
  const float* ap = agg + (size_t)n * F;
  const float* rp = res + (size_t)n * F;
  float* op = out + (size_t)n * F;
  if (!norm) {
    for (int f = 0; f < F; ++f) {
      float v = ap[f] + rp[f];
      if (act) v = fmaxf(v, 0.f);
      op[f] = v;
    }
    return;
  }
  float s1 = 0.f, s2 = 0.f;
  for (int f = 0; f < F; ++f) {
    float v = ap[f] + rp[f];
    if (act) v = fmaxf(v, 0.f);
    s1 += v; s2 += v * v;
  }
  float mean = s1 / F;
  float var = s2 / F - mean * mean;
  float inv = 1.f / sqrtf(var + 1e-5f);
  for (int f = 0; f < F; ++f) {
    float v = ap[f] + rp[f];
    if (act) v = fmaxf(v, 0.f);
    op[f] = (v - mean) * inv * gam[f] + bet[f];
  }
}

__device__ inline float sigmoidf_(float x) { return 1.f / (1.f + __expf(-x)); }

// GRU combine (in-place over hprev is safe: elementwise)
__global__ void k_gru(const float* __restrict__ gx, const float* __restrict__ gh,
                      const float* __restrict__ bx, const float* __restrict__ bh,
                      const float* __restrict__ hprev, float* __restrict__ hout,
                      int Nn, int hd) {
  int i = blockIdx.x * blockDim.x + threadIdx.x;
  if (i >= Nn * hd) return;
  int n = i / hd, j = i % hd;
  size_t b = (size_t)n * 3 * hd;
  float r  = sigmoidf_(gx[b + j]          + bx[j]          + gh[b + j]          + bh[j]);
  float z  = sigmoidf_(gx[b + hd + j]     + bx[hd + j]     + gh[b + hd + j]     + bh[hd + j]);
  float ng = tanhf(    gx[b + 2 * hd + j] + bx[2 * hd + j] + r * (gh[b + 2 * hd + j] + bh[2 * hd + j]));
  hout[i] = (1.f - z) * ng + z * hprev[i];
}

__global__ void k_concat2(const float* __restrict__ a, const float* __restrict__ b,
                          float* __restrict__ c, int Nn, int F) {
  int i = blockIdx.x * blockDim.x + threadIdx.x;
  if (i >= Nn * 2 * F) return;
  int n = i / (2 * F), f = i % (2 * F);
  c[i] = (f < F) ? a[(size_t)n * F + f] : b[(size_t)n * F + (f - F)];
}

__global__ void k_loss(const float* __restrict__ recon, const float* __restrict__ x0,
                       float* __restrict__ out) {
  int n = blockIdx.x * blockDim.x + threadIdx.x;
  if (n >= N_NODES) return;
  const float* a = recon + (size_t)n * N_DIM;
  const float* b = x0 + (size_t)n * N_DIM;
  float dot = 0.f, na = 0.f, nb = 0.f;
  for (int f = 0; f < N_DIM; ++f) { dot += a[f] * b[f]; na += a[f] * a[f]; nb += b[f] * b[f]; }
  float cosv = dot / ((sqrtf(na) + 1e-8f) * (sqrtf(nb) + 1e-8f));
  float t = 1.f - cosv;
  atomicAdd(out, (t * t) / (float)N_NODES);
}

__global__ void k_zero1(float* p) { if (threadIdx.x == 0 && blockIdx.x == 0) p[0] = 0.f; }

// ---------------------------- host orchestration ---------------------------

static inline int cdiv(long long a, int b) { return (int)((a + b - 1) / b); }
static inline void fill(float* p, float v, long long n, hipStream_t s) {
  k_fill<<<cdiv(n, 256), 256, 0, s>>>(p, v, (int)n);
}

struct Ws {
  float *emb[6], *hs0, *hs1, *dec_emb[3], *hfeat, *agg, *resb, *catb, *decin;
  float *gx, *gh, *el, *er, *m, *den, *ee, *esc, *vee, *hdec;
  _Float16* bpack;
};

static inline void gemm(const float* A, const float* B, float* C,
                        int M, int K, int N, Ws& w, hipStream_t s) {
  k_packB<<<cdiv((long long)K * N, 256), 256, 0, s>>>(B, w.bpack, K, N);
  dim3 grid(cdiv(M, 64), N / 64);
  gemm_wmma_f16<<<grid, 128, 0, s>>>(A, w.bpack, C, M, K, N);
}

static void gat_layer(hipStream_t s, Ws& w,
                      const float* x, int Fin,
                      const float* ea, const int* src, const int* dst,
                      const float* W, const float* We, const float* ae,
                      const float* al, const float* ar,
                      const float* Wres,                   // null => identity residual
                      const float* gam, const float* bet,  // null => no norm
                      int H, int dh, float* out, int act, int norm) {
  const int F = H * dh;
  gemm(x, W, w.hfeat, N_NODES, Fin, F, w, s);                       // h = x @ W
  k_vee<<<cdiv(E_DIM * H, 128), 128, 0, s>>>(We, ae, w.vee, H, dh); // vee = We @ ae
  k_node_scores<<<cdiv((long long)N_NODES * H, 256), 256, 0, s>>>(w.hfeat, al, ar, w.el, w.er, N_NODES, H, dh);
  k_ee<<<cdiv(N_EDGES, 256), 256, 0, s>>>(ea, w.vee, w.ee, H);
  fill(w.m, -INFINITY, (long long)N_NODES * H, s);
  fill(w.den, 0.f, (long long)N_NODES * H, s);
  fill(w.agg, 0.f, (long long)N_NODES * F, s);
  k_edge_score<<<cdiv((long long)N_EDGES * H, 256), 256, 0, s>>>(src, dst, w.el, w.er, w.ee, w.esc, w.m, H);
  k_clamp_m<<<cdiv((long long)N_NODES * H, 256), 256, 0, s>>>(w.m, N_NODES * H);
  k_edge_exp<<<cdiv((long long)N_EDGES * H, 256), 256, 0, s>>>(dst, w.m, w.esc, w.den, H);
  k_edge_scatter<<<cdiv((long long)N_EDGES * F, 256), 256, 0, s>>>(src, dst, w.esc, w.den, w.hfeat, w.agg, H, dh);
  const float* res = x;
  if (Wres) { gemm(x, Wres, w.resb, N_NODES, Fin, F, w, s); res = w.resb; }
  k_gat_epilogue<<<cdiv(N_NODES, 128), 128, 0, s>>>(w.agg, res, gam, bet, out, N_NODES, F, act, norm);
}

static void gru_step(hipStream_t s, Ws& w, const float* x, int ind,
                     const float* Wh, const float* Wx, const float* bh, const float* bx,
                     float* h, int hd) {
  gemm(x, Wx, w.gx, N_NODES, ind, 3 * hd, w, s);
  gemm(h, Wh, w.gh, N_NODES, hd, 3 * hd, w, s);
  k_gru<<<cdiv((long long)N_NODES * hd, 256), 256, 0, s>>>(w.gx, w.gh, bx, bh, h, h, N_NODES, hd);
}

extern "C" void kernel_launch(void* const* d_in, const int* in_sizes, int n_in,
                              void* d_out, int out_size, void* d_ws, size_t ws_size,
                              hipStream_t stream) {
  (void)in_sizes; (void)n_in; (void)out_size; (void)ws_size;
  const float* node_feats = (const float*)d_in[0];   // [3,N,128]
  const float* edge_attr  = (const float*)d_in[1];   // [3,E,32]
  const int*   esrc       = (const int*)  d_in[2];   // [3,E]
  const int*   edst       = (const int*)  d_in[3];   // [3,E]
  // params flattened in jax pytree (sorted-key) order:
  const float* dW    = (const float*)d_in[4];   // dec_gat.W   (64,1,128)
  const float* dWe   = (const float*)d_in[5];   // dec_gat.We  (32,1,128)
  const float* dWres = (const float*)d_in[6];   // dec_gat.Wres(64,128)
  const float* dae   = (const float*)d_in[7];
  const float* dal   = (const float*)d_in[8];
  const float* dar   = (const float*)d_in[9];
  // dec_rnn[c]: Wh, Wx, bh, bx @ 10+4c
  const float* e2d   = (const float*)d_in[22];  // (128,64)
  const float* g0W    = (const float*)d_in[23];
  const float* g0We   = (const float*)d_in[24];
  const float* g0Wres = (const float*)d_in[25];
  const float* g0ae   = (const float*)d_in[26];
  const float* g0al   = (const float*)d_in[27];
  const float* g0ar   = (const float*)d_in[28];
  const float* g0b    = (const float*)d_in[29];
  const float* g0g    = (const float*)d_in[30];
  const float* g1W    = (const float*)d_in[31];
  const float* g1We   = (const float*)d_in[32];
  const float* g1ae   = (const float*)d_in[33];
  const float* g1al   = (const float*)d_in[34];
  const float* g1ar   = (const float*)d_in[35];
  const float* g1b    = (const float*)d_in[36];
  const float* g1g    = (const float*)d_in[37];
  // enc_rnn[c]: Wh, Wx, bh, bx @ 38+4c
  float* out = (float*)d_out;

  // workspace carve-up
  Ws w; {
    float* p = (float*)d_ws;
    size_t off = 0;
    auto al_ = [&](size_t n) { float* q = p + off; off += n; return q; };
    for (int i = 0; i < 6; ++i) w.emb[i] = al_((size_t)N_NODES * 64);
    w.hs0 = al_((size_t)N_NODES * 64);
    w.hs1 = al_((size_t)N_NODES * 64);
    for (int i = 0; i < 3; ++i) w.dec_emb[i] = al_((size_t)N_NODES * 128);
    w.hfeat = al_((size_t)N_NODES * 128);
    w.agg   = al_((size_t)N_NODES * 128);
    w.resb  = al_((size_t)N_NODES * 128);
    w.catb  = al_((size_t)N_NODES * 128);
    w.decin = al_((size_t)N_NODES * 64);
    w.gx    = al_((size_t)N_NODES * 384);
    w.gh    = al_((size_t)N_NODES * 384);
    w.el    = al_((size_t)N_NODES * HEADS);
    w.er    = al_((size_t)N_NODES * HEADS);
    w.m     = al_((size_t)N_NODES * HEADS);
    w.den   = al_((size_t)N_NODES * HEADS);
    w.ee    = al_((size_t)N_EDGES * HEADS);
    w.esc   = al_((size_t)N_EDGES * HEADS);
    w.vee   = al_(E_DIM * HEADS);
    w.hdec  = al_((size_t)N_NODES * 128);
    w.bpack = (_Float16*)al_(128 * 384 / 2 + 128);  // max K*N halves
  }

  // ---------------- encoder GAT (2 layers x 3 snapshots) -------------------
  for (int i = 0; i < 3; ++i) {
    const float* x  = node_feats + (size_t)i * N_NODES * N_DIM;
    const float* ea = edge_attr + (size_t)i * N_EDGES * E_DIM;
    const int* sp = esrc + (size_t)i * N_EDGES;
    const int* dp = edst + (size_t)i * N_EDGES;
    gat_layer(stream, w, x, N_DIM, ea, sp, dp,
              g0W, g0We, g0ae, g0al, g0ar, g0Wres, g0g, g0b,
              HEADS, OUT_DIM / HEADS, w.emb[2 * i + 0], 1, 1);
    gat_layer(stream, w, w.emb[2 * i + 0], OUT_DIM, ea, sp, dp,
              g1W, g1We, g1ae, g1al, g1ar, nullptr, g1g, g1b,
              HEADS, OUT_DIM / HEADS, w.emb[2 * i + 1], 1, 1);
  }

  // ---------------- encoder GRUs (2 streams), e2d, decoder GAT -------------
  fill(w.hs0, 0.f, (size_t)N_NODES * 64, stream);
  fill(w.hs1, 0.f, (size_t)N_NODES * 64, stream);
  for (int i = 0; i < 3; ++i) {
    const float* Wh = (const float*)d_in[38 + 4 * i];
    const float* Wx = (const float*)d_in[39 + 4 * i];
    const float* bh = (const float*)d_in[40 + 4 * i];
    const float* bx = (const float*)d_in[41 + 4 * i];
    gru_step(stream, w, w.emb[2 * i + 0], 64, Wh, Wx, bh, bx, w.hs0, 64);
    gru_step(stream, w, w.emb[2 * i + 1], 64, Wh, Wx, bh, bx, w.hs1, 64);
    k_concat2<<<cdiv((long long)N_NODES * 128, 256), 256, 0, stream>>>(w.hs0, w.hs1, w.catb, N_NODES, 64);
    gemm(w.catb, e2d, w.decin, N_NODES, 128, 64, w, stream);
    const float* ea = edge_attr + (size_t)i * N_EDGES * E_DIM;
    const int* sp = esrc + (size_t)i * N_EDGES;
    const int* dp = edst + (size_t)i * N_EDGES;
    gat_layer(stream, w, w.decin, 64, ea, sp, dp,
              dW, dWe, dae, dal, dar, dWres, nullptr, nullptr,
              1, N_DIM, w.dec_emb[i], 0, 0);
  }

  // ---------------- decoder GRU over snapshots -----------------------------
  fill(w.hdec, 0.f, (size_t)N_NODES * 128, stream);
  for (int i = 0; i < 3; ++i) {
    const float* Wh = (const float*)d_in[10 + 4 * i];
    const float* Wx = (const float*)d_in[11 + 4 * i];
    const float* bh = (const float*)d_in[12 + 4 * i];
    const float* bx = (const float*)d_in[13 + 4 * i];
    gru_step(stream, w, w.dec_emb[i], 128, Wh, Wx, bh, bx, w.hdec, 128);
  }

  // ---------------- SCE loss vs snapshot-0 features ------------------------
  k_zero1<<<1, 32, 0, stream>>>(out);
  k_loss<<<cdiv(N_NODES, 256), 256, 0, stream>>>(w.hdec, node_feats, out);
}
